// Block_49151605735799
// MI455X (gfx1250) — compile-verified, matrix operand order checked by reading, not run on hardware
//
#include <hip/hip_runtime.h>
#include <cstdint>
#include <cstddef>

// ---------------------------------------------------------------------------
// Transformer block (B=2, T=2048, C=1024, H=16, D=64) for gfx1250 (CDNA5).
// All GEMMs + attention use V_WMMA_F32_16X16X32_BF16 (bf16 in, f32 accum).
// GEMM: 128x128 block tile, 4 waves (64x64 wave tile), K-step 32,
// double-buffered GLOBAL_LOAD_ASYNC_TO_LDS_B128 staging (ASYNCcnt).
// Weights pre-converted/pre-transposed to bf16; QKV fused into one GEMM.
// ---------------------------------------------------------------------------

#define TB 2
#define TT 2048
#define TC 1024
#define TH 16
#define TD 64
#define TM (TB * TT)   // 4096 rows of activations

typedef __bf16 bf16_t;
typedef bf16_t v16bf __attribute__((ext_vector_type(16)));
typedef float  v8f   __attribute__((ext_vector_type(8)));

struct alignas(16) U128 { uint32_t w[4]; };

union FragBF {
    v16bf v;
    U128  u[2];
    __device__ FragBF() {}
};

__device__ __forceinline__ unsigned short f2bf(float f) {
    union { float f; uint32_t u; } x; x.f = f;
    uint32_t r = x.u + 0x7FFFu + ((x.u >> 16) & 1u);   // round-to-nearest-even
    return (unsigned short)(r >> 16);
}

__device__ __forceinline__ v8f vzero8() {
    v8f z;
#pragma unroll
    for (int i = 0; i < 8; ++i) z[i] = 0.0f;
    return z;
}

// Async global->LDS 16-byte copy (CDNA5), tracked by ASYNCcnt.
__device__ __forceinline__ void async_copy_b128(uint32_t lds_off,
                                                const unsigned short* gptr) {
    asm volatile("global_load_async_to_lds_b128 %0, %1, off"
                 :: "v"(lds_off), "v"(gptr)
                 : "memory");
}
__device__ __forceinline__ void wait_async0() {
    asm volatile("s_wait_asynccnt 0x0" ::: "memory");
}

// ---------------------------------------------------------------------------
// Weight prepass: W f32 [K,N] -> Wt bf16 [N,K] (convert + transpose, once).
// Block handles a 32x32 tile; grid = (N/32, K/32).
// ---------------------------------------------------------------------------
__global__ __launch_bounds__(256)
void wconv_kernel(const float* __restrict__ W, unsigned short* __restrict__ Wt,
                  int K, int N) {
    __shared__ float tile[32][33];
    const int k0 = blockIdx.y * 32, n0 = blockIdx.x * 32;
    const int tid = threadIdx.x;
    const int c = tid & 31, rg = tid >> 5;   // 8 rows per pass
#pragma unroll
    for (int i = 0; i < 4; ++i) {
        int kk = rg + i * 8;
        tile[kk][c] = W[(size_t)(k0 + kk) * N + n0 + c];
    }
    __syncthreads();
#pragma unroll
    for (int i = 0; i < 4; ++i) {
        int nn = rg + i * 8;
        Wt[(size_t)(n0 + nn) * K + k0 + c] = f2bf(tile[c][nn]);
    }
}

// Concatenate three length-TC bias vectors into one length-3*TC vector.
__global__ __launch_bounds__(256)
void bcat_kernel(const float* __restrict__ b0, const float* __restrict__ b1,
                 const float* __restrict__ b2, float* __restrict__ out) {
    int i = blockIdx.x * 256 + threadIdx.x;   // 0..3071
    const float* src = (i < TC) ? b0 : (i < 2 * TC) ? b1 : b2;
    out[i] = src[i & (TC - 1)];
}

// ---------------------------------------------------------------------------
// LayerNorm: f32 [rows, C] -> bf16 [rows, C]
// ---------------------------------------------------------------------------
__global__ __launch_bounds__(256)
void ln_kernel(const float* __restrict__ x, const float* __restrict__ w,
               const float* __restrict__ b, unsigned short* __restrict__ out) {
    const int C = TC;
    const int row = blockIdx.x;
    const float* xr = x + (size_t)row * C;
    float s = 0.f, s2 = 0.f;
    for (int i = threadIdx.x; i < C; i += 256) {
        float v = xr[i]; s += v; s2 += v * v;
    }
#pragma unroll
    for (int m = 16; m >= 1; m >>= 1) {
        s  += __shfl_xor(s,  m, 32);
        s2 += __shfl_xor(s2, m, 32);
    }
    __shared__ float rs[8], rs2[8];
    int wv = threadIdx.x >> 5, lane = threadIdx.x & 31;
    if (lane == 0) { rs[wv] = s; rs2[wv] = s2; }
    __syncthreads();
    if (wv == 0) {
        s  = (lane < 8) ? rs[lane]  : 0.f;
        s2 = (lane < 8) ? rs2[lane] : 0.f;
#pragma unroll
        for (int m = 4; m >= 1; m >>= 1) {
            s  += __shfl_xor(s,  m, 32);
            s2 += __shfl_xor(s2, m, 32);
        }
        if (lane == 0) { rs[0] = s; rs2[0] = s2; }
    }
    __syncthreads();
    float mu  = rs[0] / (float)C;
    float var = rs2[0] / (float)C - mu * mu;
    float inv = rsqrtf(var + 1e-5f);
    for (int i = threadIdx.x; i < C; i += 256) {
        float v = (xr[i] - mu) * inv * w[i] + b[i];
        out[(size_t)row * C + i] = f2bf(v);
    }
}

// ---------------------------------------------------------------------------
// WMMA GEMM: out[M,N] = A_bf16[M,K] @ Wt_bf16[N,K]^T + bias, optional GELU,
// optional residual, f32 and/or bf16 out (bf16 optionally head-scattered;
// SCATTER also demuxes fused-QKV columns into contiguous q/k/v buffers).
// Block tile 128x128, 128 threads = 4 waves (2x2), wave tile 64x64.
// K-step 32. Double-buffered async LDS staging.
// ---------------------------------------------------------------------------
#define GEMM_LDS_STRIDE 40                            // ushorts per row
#define GEMM_TILE_BYTES (128 * GEMM_LDS_STRIDE * 2)   // 10240
#define GEMM_BUF_BYTES  (2 * GEMM_TILE_BYTES)         // A + B
#define GEMM_SMEM_BYTES (2 * GEMM_BUF_BYTES)          // double buffered: 40960

template <bool GELU, bool SCATTER>
__global__ __launch_bounds__(128)
void gemm_kernel(const unsigned short* __restrict__ A,
                 const unsigned short* __restrict__ Wt,
                 const float* __restrict__ bias,
                 const float* __restrict__ residual,
                 float* __restrict__ outF,
                 unsigned short* __restrict__ outB,
                 int M, int N, int K) {
    extern __shared__ char smem[];               // dynamic LDS: base offset 0

    const int m0 = blockIdx.y * 128;
    const int n0 = blockIdx.x * 128;
    const int tid = threadIdx.x;
    const int lane = tid & 31;
    const int wv = tid >> 5;
    const int wm = (wv >> 1) * 64;               // 2 wave rows
    const int wn = (wv & 1) * 64;                // 2 wave cols
    const int m16 = lane & 15;
    const int g = lane >> 4;

    v8f acc[4][4];
#pragma unroll
    for (int i = 0; i < 4; ++i)
#pragma unroll
        for (int j = 0; j < 4; ++j) acc[i][j] = vzero8();

    // Stage one 128x32 A tile + 128x32 B tile into LDS buffer `buf` (async).
    // 128 threads x (4 A + 4 B) 16-byte copies.
    auto stage = [&](int buf, int k0) {
        uint32_t base = (uint32_t)buf * GEMM_BUF_BYTES;
        const unsigned short* ga = A  + (size_t)(m0 + tid) * K + k0;
        const unsigned short* gb = Wt + (size_t)(n0 + tid) * K + k0;
        uint32_t off = base + (uint32_t)tid * (GEMM_LDS_STRIDE * 2);
#pragma unroll
        for (int q = 0; q < 4; ++q) {
            async_copy_b128(off + q * 16, ga + q * 8);
            async_copy_b128(off + GEMM_TILE_BYTES + q * 16, gb + q * 8);
        }
    };

    stage(0, 0);
    int buf = 0;
    for (int k0 = 0; k0 < K; k0 += 32) {
        wait_async0();
        __syncthreads();
        if (k0 + 32 < K) stage(buf ^ 1, k0 + 32);   // overlap with compute

        const unsigned short* lA =
            (const unsigned short*)(smem + buf * GEMM_BUF_BYTES);
        const unsigned short* lB =
            (const unsigned short*)(smem + buf * GEMM_BUF_BYTES + GEMM_TILE_BYTES);

        FragBF a[4];
#pragma unroll
        for (int i = 0; i < 4; ++i) {
            int row = wm + i * 16 + m16;
            a[i].u[0] = *(const U128*)(&lA[row * GEMM_LDS_STRIDE + g * 8]);
            a[i].u[1] = *(const U128*)(&lA[row * GEMM_LDS_STRIDE + 16 + g * 8]);
        }
#pragma unroll
        for (int j = 0; j < 4; ++j) {
            FragBF b;
            int row = wn + j * 16 + m16;
            b.u[0] = *(const U128*)(&lB[row * GEMM_LDS_STRIDE + g * 8]);
            b.u[1] = *(const U128*)(&lB[row * GEMM_LDS_STRIDE + 16 + g * 8]);
#pragma unroll
            for (int i = 0; i < 4; ++i)
                acc[i][j] = __builtin_amdgcn_wmma_f32_16x16x32_bf16(
                    false, a[i].v, false, b.v, (short)0, acc[i][j], false, false);
        }
        buf ^= 1;
    }

    // --- epilogue ---
#pragma unroll
    for (int i = 0; i < 4; ++i)
#pragma unroll
        for (int j = 0; j < 4; ++j)
#pragma unroll
            for (int r = 0; r < 8; ++r) {
                int grow = m0 + wm + i * 16 + r + 8 * g;
                int gcol = n0 + wn + j * 16 + m16;
                float v = acc[i][j][r] + bias[gcol];
                if (GELU) v = 0.5f * v * (1.0f + erff(v * 0.70710678118654752f));
                if (residual) v += residual[(size_t)grow * N + gcol];
                if (outF) outF[(size_t)grow * N + gcol] = v;
                if (outB) {
                    size_t oi;
                    if (SCATTER) {
                        // fused QKV demux: matrix = gcol/1024, then head layout
                        int mat = gcol >> 10, wc = gcol & (TC - 1);
                        int bb = grow >> 11, t = grow & (TT - 1);
                        int hh = wc >> 6,    d = wc & (TD - 1);
                        oi = (size_t)mat * ((size_t)TM * TC) +
                             (((size_t)(bb * TH + hh)) * TT + t) * TD + d;
                    } else {
                        oi = (size_t)grow * N + gcol;
                    }
                    outB[oi] = f2bf(v);
                }
            }
}

// ---------------------------------------------------------------------------
// Flash attention: one wave per (b,h, 16-row q block). Q,K,V bf16 [B*H,T,D].
// Output written as bf16 activation [B*T, C].
// ---------------------------------------------------------------------------
__global__ __launch_bounds__(32)
void attn_kernel(const unsigned short* __restrict__ Q,
                 const unsigned short* __restrict__ K,
                 const unsigned short* __restrict__ V,
                 unsigned short* __restrict__ O) {
    const int bid = blockIdx.x;
    const int qb = bid & 127;     // T/16 = 128 q-blocks
    const int bh = bid >> 7;      // 0..31
    const int q0 = qb * 16;
    const int lane = threadIdx.x;
    const int m16 = lane & 15;
    const int g = lane >> 4;

    const unsigned short* Qb = Q + (size_t)bh * TT * TD;
    const unsigned short* Kb = K + (size_t)bh * TT * TD;
    const unsigned short* Vb = V + (size_t)bh * TT * TD;

    __shared__ unsigned short lV[64 * 40];  // Vt[d][t] for current 32-key chunk
    __shared__ unsigned short lP[16 * 40];  // P tile [q][t]

    // Q fragments: A-layout, loaded straight from global (row-major [t][d])
    FragBF qf[2];
#pragma unroll
    for (int c = 0; c < 2; ++c) {
        const unsigned short* p = Qb + (size_t)(q0 + m16) * TD + c * 32;
        qf[c].u[0] = *(const U128*)(p + g * 8);
        qf[c].u[1] = *(const U128*)(p + 16 + g * 8);
    }

    v8f o[4];
#pragma unroll
    for (int j = 0; j < 4; ++j) o[j] = vzero8();
    float mprev[8], lsum[8];
#pragma unroll
    for (int r = 0; r < 8; ++r) { mprev[r] = -3e38f; lsum[r] = 0.f; }

    const int nch = (q0 + 15) / 32 + 1;
    for (int c = 0; c < nch; ++c) {
        const int t0 = c * 32;
        // prefetch next chunk of K and V into near caches
        if (c + 1 < nch) {
            __builtin_prefetch(Kb + (size_t)(t0 + 32 + lane) * TD, 0, 3);
            __builtin_prefetch(Vb + (size_t)(t0 + 32 + lane) * TD, 0, 3);
        }
        // load all 4 K fragments first, then issue the 4 WMMAs
        FragBF kf[2][2];
#pragma unroll
        for (int ks = 0; ks < 2; ++ks)
#pragma unroll
            for (int dc = 0; dc < 2; ++dc) {
                const unsigned short* p =
                    Kb + (size_t)(t0 + ks * 16 + m16) * TD + dc * 32;
                kf[ks][dc].u[0] = *(const U128*)(p + g * 8);
                kf[ks][dc].u[1] = *(const U128*)(p + 16 + g * 8);
            }
        v8f s0 = vzero8(), s1 = vzero8();
#pragma unroll
        for (int dc = 0; dc < 2; ++dc)
            s0 = __builtin_amdgcn_wmma_f32_16x16x32_bf16(
                false, qf[dc].v, false, kf[0][dc].v, (short)0, s0, false, false);
#pragma unroll
        for (int dc = 0; dc < 2; ++dc)
            s1 = __builtin_amdgcn_wmma_f32_16x16x32_bf16(
                false, qf[dc].v, false, kf[1][dc].v, (short)0, s1, false, false);

        const float sc = 0.125f;  // 1/sqrt(64)
#pragma unroll
        for (int r = 0; r < 8; ++r) { s0[r] *= sc; s1[r] *= sc; }
        if (t0 + 31 > q0) {       // causal mask on diagonal / beyond chunks
#pragma unroll
            for (int r = 0; r < 8; ++r) {
                int qrow = q0 + r + 8 * g;
                if (t0 + m16 > qrow)      s0[r] = -3e38f;
                if (t0 + 16 + m16 > qrow) s1[r] = -3e38f;
            }
        }
        // online softmax: row stats live in C-layout halves
        float mnew[8], alpha[8], lblk[8];
#pragma unroll
        for (int r = 0; r < 8; ++r) {
            float mb = fmaxf(s0[r], s1[r]);
#pragma unroll
            for (int msk = 1; msk < 16; msk <<= 1)
                mb = fmaxf(mb, __shfl_xor(mb, msk, 32));
            mnew[r] = fmaxf(mprev[r], mb);
            alpha[r] = __expf(mprev[r] - mnew[r]);
        }
#pragma unroll
        for (int r = 0; r < 8; ++r) {
            float p0 = __expf(s0[r] - mnew[r]);
            float p1 = __expf(s1[r] - mnew[r]);
            int row = r + 8 * g;
            lP[row * 40 + m16]      = f2bf(p0);
            lP[row * 40 + 16 + m16] = f2bf(p1);
            float t = p0 + p1;
#pragma unroll
            for (int msk = 1; msk < 16; msk <<= 1)
                t += __shfl_xor(t, msk, 32);
            lblk[r] = t;
        }
#pragma unroll
        for (int r = 0; r < 8; ++r) {
            lsum[r] = lsum[r] * alpha[r] + lblk[r];
            mprev[r] = mnew[r];
        }
        // stage V chunk transposed: lV[d][t]
        for (int i = 0; i < 64; ++i) {
            int idx = i * 32 + lane;
            int tt = idx >> 6, dd = idx & 63;
            lV[dd * 40 + tt] = Vb[(size_t)(t0 + tt) * TD + dd];
        }
        __syncthreads();
        // P fragment (A-layout) from LDS
        FragBF pf;
        pf.u[0] = *(const U128*)(&lP[m16 * 40 + g * 8]);
        pf.u[1] = *(const U128*)(&lP[m16 * 40 + 16 + g * 8]);
        // O = O*alpha + P @ V
#pragma unroll
        for (int j = 0; j < 4; ++j) {
#pragma unroll
            for (int r = 0; r < 8; ++r) o[j][r] *= alpha[r];
            FragBF vf;
            vf.u[0] = *(const U128*)(&lV[(j * 16 + m16) * 40 + g * 8]);
            vf.u[1] = *(const U128*)(&lV[(j * 16 + m16) * 40 + 16 + g * 8]);
            o[j] = __builtin_amdgcn_wmma_f32_16x16x32_bf16(
                false, pf.v, false, vf.v, (short)0, o[j], false, false);
        }
        __syncthreads();
    }

    // finalize: O /= l, write bf16 activation [B*T, C]
    const int bb = bh >> 4, hh = bh & 15;
#pragma unroll
    for (int j = 0; j < 4; ++j)
#pragma unroll
        for (int r = 0; r < 8; ++r) {
            float v = o[j][r] / lsum[r];
            size_t row = (size_t)bb * TT + q0 + r + 8 * g;
            size_t col = (size_t)hh * TD + j * 16 + m16;
            O[row * TC + col] = f2bf(v);
        }
}

// ---------------------------------------------------------------------------
// Launch
// ---------------------------------------------------------------------------
extern "C" void kernel_launch(void* const* d_in, const int* in_sizes, int n_in,
                              void* d_out, int out_size, void* d_ws, size_t ws_size,
                              hipStream_t stream) {
    const float* x     = (const float*)d_in[0];
    const float* ln1_w = (const float*)d_in[1];
    const float* ln1_b = (const float*)d_in[2];
    const float* wq    = (const float*)d_in[3];
    const float* bq    = (const float*)d_in[4];
    const float* wk    = (const float*)d_in[5];
    const float* bk    = (const float*)d_in[6];
    const float* wv    = (const float*)d_in[7];
    const float* bv    = (const float*)d_in[8];
    const float* wo    = (const float*)d_in[9];
    const float* bo    = (const float*)d_in[10];
    const float* ln2_w = (const float*)d_in[11];
    const float* ln2_b = (const float*)d_in[12];
    const float* w1    = (const float*)d_in[13];
    const float* b1    = (const float*)d_in[14];
    const float* w2    = (const float*)d_in[15];
    const float* b2    = (const float*)d_in[16];
    float* out = (float*)d_out;

    char* ws = (char*)d_ws;
    const size_t MB = 1024 * 1024;
    unsigned short* h1   = (unsigned short*)(ws);              //  8 MB, reused as h2
    unsigned short* qbuf = (unsigned short*)(ws + 8  * MB);    //  8 MB  (q|k|v contiguous)
    unsigned short* kbuf = (unsigned short*)(ws + 16 * MB);    //  8 MB
    unsigned short* vbuf = (unsigned short*)(ws + 24 * MB);    //  8 MB
    unsigned short* abuf = (unsigned short*)(ws + 32 * MB);    //  8 MB
    unsigned short* gbuf = (unsigned short*)(ws + 8  * MB);    // 32 MB, reuses q/k/v/attn
    float*          x2   = (float*)(ws + 40 * MB);             // 16 MB
    unsigned short* h2   = h1;
    unsigned short* wqT  = (unsigned short*)(ws + 56 * MB);    // 2 MB (q|k|v contiguous)
    unsigned short* wkT  = (unsigned short*)(ws + 58 * MB);    // 2 MB
    unsigned short* wvT  = (unsigned short*)(ws + 60 * MB);    // 2 MB
    unsigned short* woT  = (unsigned short*)(ws + 62 * MB);    // 2 MB
    unsigned short* w1T  = (unsigned short*)(ws + 64 * MB);    // 8 MB
    unsigned short* w2T  = (unsigned short*)(ws + 72 * MB);    // 8 MB
    float*          bqkv = (float*)(ws + 80 * MB);             // 12 KB

    // 0) one-time weight convert+transpose (f32 [K,N] -> bf16 [N,K]) + bias cat
    wconv_kernel<<<dim3(TC / 32, TC / 32), 256, 0, stream>>>(wq, wqT, TC, TC);
    wconv_kernel<<<dim3(TC / 32, TC / 32), 256, 0, stream>>>(wk, wkT, TC, TC);
    wconv_kernel<<<dim3(TC / 32, TC / 32), 256, 0, stream>>>(wv, wvT, TC, TC);
    wconv_kernel<<<dim3(TC / 32, TC / 32), 256, 0, stream>>>(wo, woT, TC, TC);
    wconv_kernel<<<dim3(4 * TC / 32, TC / 32), 256, 0, stream>>>(w1, w1T, TC, 4 * TC);
    wconv_kernel<<<dim3(TC / 32, 4 * TC / 32), 256, 0, stream>>>(w2, w2T, 4 * TC, TC);
    bcat_kernel<<<12, 256, 0, stream>>>(bq, bk, bv, bqkv);

    const dim3 gemm_block(128);
    const dim3 grid_cc(TC / 128, TM / 128);       // N=1024
    const dim3 grid_c3(3 * TC / 128, TM / 128);   // N=3072 (fused QKV)
    const dim3 grid_c4(4 * TC / 128, TM / 128);   // N=4096

    // 1) h1 = LN(x)
    ln_kernel<<<TM, 256, 0, stream>>>(x, ln1_w, ln1_b, h1);

    // 2) fused q|k|v = h1 @ [wq|wk|wv] + [bq|bk|bv], scattered to [B,H,T,D] bf16
    gemm_kernel<false, true><<<grid_c3, gemm_block, GEMM_SMEM_BYTES, stream>>>(
        h1, wqT, bqkv, nullptr, nullptr, qbuf, TM, 3 * TC, TC);

    // 3) causal flash attention -> abuf [B*T, C] bf16
    attn_kernel<<<TB * TH * (TT / 16), 32, 0, stream>>>(qbuf, kbuf, vbuf, abuf);

    // 4) x2 = x + abuf @ wo + bo
    gemm_kernel<false, false><<<grid_cc, gemm_block, GEMM_SMEM_BYTES, stream>>>(
        abuf, woT, bo, x, x2, nullptr, TM, TC, TC);

    // 5) h2 = LN(x2)
    ln_kernel<<<TM, 256, 0, stream>>>(x2, ln2_w, ln2_b, h2);

    // 6) g = gelu(h2 @ w1 + b1)  (bf16)
    gemm_kernel<true, false><<<grid_c4, gemm_block, GEMM_SMEM_BYTES, stream>>>(
        h2, w1T, b1, nullptr, nullptr, gbuf, TM, 4 * TC, TC);

    // 7) out = x2 + g @ w2 + b2
    gemm_kernel<false, false><<<grid_cc, gemm_block, GEMM_SMEM_BYTES, stream>>>(
        gbuf, w2T, b2, x2, out, nullptr, TM, TC, 4 * TC);

    (void)in_sizes; (void)n_in; (void)out_size; (void)ws_size;
    (void)kbuf; (void)vbuf;
}
